// EnhancedSTGCNLayer_21500606284427
// MI455X (gfx1250) — matrix operand order repaired and checked by reading, not compiled, not run on hardware
//
#include <hip/hip_runtime.h>
#include <hip/hip_bf16.h>
#include <math.h>

constexpr int Nn   = 20000;
constexpr int Ee   = 160000;
constexpr int Tt   = 8;
constexpr int CI   = 64;
constexpr int CO   = 128;

typedef _Float16 v16h __attribute__((ext_vector_type(16)));
typedef _Float16 v8h  __attribute__((ext_vector_type(8)));
typedef float    v8f  __attribute__((ext_vector_type(8)));
typedef unsigned int u32;
typedef u32 tdmg0 __attribute__((ext_vector_type(4)));
typedef int tdmg1 __attribute__((ext_vector_type(8)));
typedef int tdmg2 __attribute__((ext_vector_type(4)));
typedef int tdmg4 __attribute__((ext_vector_type(8)));

#define WMMA16(A,B,C) __builtin_amdgcn_wmma_f32_16x16x32_f16(false,(A),false,(B),(short)0,(C),false,false)

// ---- workspace layout (float units) ----
constexpr size_t OFF_EW   = 0;                       // E floats: softmaxed edge weights
constexpr size_t OFF_DEG  = OFF_EW + Ee;             // N floats: degree -> dinv (in place)
constexpr size_t OFF_SC   = OFF_DEG + Nn;            // [0]=max, [1]=sum
constexpr size_t OFF_BLOB = OFF_SC + 4;              // 96*512 halves, 16B-aligned
constexpr size_t BLOB_H   = 96 * 512;
constexpr size_t OFF_XT   = OFF_BLOB + BLOB_H/2;     // N*T*CO floats

// ------------------------------------------------------------------ init
__global__ void k_init(float* __restrict__ deg, float* __restrict__ sc) {
  int i = blockIdx.x * blockDim.x + threadIdx.x;
  if (i < Nn) deg[i] = 1.0f;                         // self-loop weight
  if (i == 0) { sc[0] = -3.4e38f; sc[1] = 0.f; }
}

__device__ __forceinline__ void atomicMaxF(float* a, float v) {
  int* ai = (int*)a;
  int old = *ai;
  while (__int_as_float(old) < v) {
    int assumed = old;
    old = atomicCAS(ai, assumed, __float_as_int(v));
    if (old == assumed) break;
  }
}

// ------------------------------------------------------------ softmax max
__global__ void k_max(const float* __restrict__ w, float* __restrict__ sc) {
  __shared__ float red[256];
  float m = -3.4e38f;
  for (int i = blockIdx.x * 256 + threadIdx.x; i < Ee; i += gridDim.x * 256)
    m = fmaxf(m, w[i]);
  red[threadIdx.x] = m; __syncthreads();
  for (int k = 128; k > 0; k >>= 1) {
    if (threadIdx.x < k) red[threadIdx.x] = fmaxf(red[threadIdx.x], red[threadIdx.x + k]);
    __syncthreads();
  }
  if (threadIdx.x == 0) atomicMaxF(&sc[0], red[0]);
}

// ------------------------------------------------------------ softmax sum
__global__ void k_sum(const float* __restrict__ w, float* __restrict__ sc) {
  __shared__ float red[256];
  const float mx = sc[0];
  float s = 0.f;
  for (int i = blockIdx.x * 256 + threadIdx.x; i < Ee; i += gridDim.x * 256)
    s += expf(w[i] - mx);
  red[threadIdx.x] = s; __syncthreads();
  for (int k = 128; k > 0; k >>= 1) {
    if (threadIdx.x < k) red[threadIdx.x] += red[threadIdx.x + k];
    __syncthreads();
  }
  if (threadIdx.x == 0) atomicAdd(&sc[1], red[0]);
}

// -------------------------------------------- per-edge weight + degree
__global__ void k_edge(const float* __restrict__ w, const int* __restrict__ ei,
                       const float* __restrict__ sc, float* __restrict__ ew,
                       float* __restrict__ deg) {
  int e = blockIdx.x * 256 + threadIdx.x;
  if (e >= Ee) return;
  float v = expf(w[e] - sc[0]) / sc[1];
  ew[e] = v;
  atomicAdd(&deg[ei[Ee + e]], v);                    // col side
}

__global__ void k_dinv(float* __restrict__ deg) {
  int n = blockIdx.x * 256 + threadIdx.x;
  if (n < Nn) deg[n] = rsqrtf(deg[n]);               // deg >= 1 always
}

// ---------------------------------------------------- weight -> f16 blob
// Fragment fi, lane l, half h:  B[K][col], K = (h<8 ? (l>>4)*8 + h : 16 + (l>>4)*8 + h-8), col = ot*16 + (l&15)
__global__ void k_wprep(const float* __restrict__ tconv_w, const float* __restrict__ res_w,
                        const float* __restrict__ gcn_w, _Float16* __restrict__ blob) {
  int idx = blockIdx.x * 256 + threadIdx.x;          // < 96*512
  int fi = idx >> 9;
  int l  = (idx >> 4) & 31;
  int h  = idx & 15;
  int lcol = l & 15;
  int kh = (h < 8) ? ((l >> 4) << 3) + h : 16 + ((l >> 4) << 3) + (h - 8); // K within 32-tile
  float v;
  if (fi < 48) {
    int s = fi >> 4, kt = (fi >> 3) & 1, ot = fi & 7;
    int c = kt * 32 + kh, o = ot * 16 + lcol;
    v = tconv_w[(o * CI + c) * 3 + s];               // [CO][CI][3]
  } else if (fi < 64) {
    int kt = (fi - 48) >> 3, ot = fi & 7;
    int c = kt * 32 + kh, o = ot * 16 + lcol;
    v = res_w[o * CI + c];                           // [CO][CI]
  } else {
    int kt = (fi - 64) >> 3, ot = fi & 7;
    int c = kt * 32 + kh, o = ot * 16 + lcol;
    v = gcn_w[o * CO + c];                           // [CO][CO]
  }
  blob[idx] = (_Float16)v;
}

// ---------------------------------------------- TDM 2-D tile load (f32 elems)
// D# per CDNA5 ISA ch.8: group0 = count/lds_addr/global_addr/type=2,
// group1 = data_size(4B), tensor dims, tile dims, dim0 stride. groups 2/3 zero (<=2D).
__device__ __forceinline__ void tdm_load_2d(u32 lds_addr, const void* gptr,
                                            u32 dim0, u32 dim1, u32 tile0, u32 tile1,
                                            u32 stride0) {
  unsigned long long ga = (unsigned long long)(uintptr_t)gptr;
  tdmg0 g0;
  g0[0] = 1u;                                        // count=1, load, user mode
  g0[1] = lds_addr;                                  // bits [63:32]
  g0[2] = (u32)(ga & 0xffffffffu);                   // global_addr[31:0]
  g0[3] = (u32)((ga >> 32) & 0x01ffffffu) | (2u << 30); // addr[56:32] | type=2
  tdmg1 g1;
  g1[0] = (int)(2u << 16);                           // data_size = 4 bytes
  g1[1] = (int)(dim0 << 16);                         // tensor_dim0[15:0]
  g1[2] = (int)((dim0 >> 16) | (dim1 << 16));        // dim0[31:16] | dim1[15:0]
  g1[3] = (int)((dim1 >> 16) | (tile0 << 16));       // dim1[31:16] | tile_dim0
  g1[4] = (int)(tile1 & 0xffffu);                    // tile_dim1 | tile_dim2=0
  g1[5] = (int)stride0;                              // tensor_dim0_stride[31:0]
  g1[6] = 0;                                         // stride hi | dim1_stride lo
  g1[7] = 0;
  tdmg2 z4 = {0, 0, 0, 0};
  tdmg4 z8 = {0, 0, 0, 0, 0, 0, 0, 0};
  __builtin_amdgcn_tensor_load_to_lds(g0, g1, z4, z4, z8, 0);
}

// ------------------------------------------------------- fused WMMA kernel
// block = 256 threads = 8 waves; wave handles 16 rows = 2 nodes x 8 timesteps
__global__ __launch_bounds__(256, 2) void kb_fused(
    const float* __restrict__ x, const _Float16* __restrict__ blob,
    const float* __restrict__ tconv_b, const float* __restrict__ ln_g,
    const float* __restrict__ ln_b, const float* __restrict__ res_b,
    float* __restrict__ xt) {
  extern __shared__ char smem[];
  _Float16* sB  = (_Float16*)smem;                   // 49152 halves (96KB)
  float* zrow   = (float*)(sB + 96 * 512);           // 128 f32 zero pad row
  float* sX     = zrow + 128;                        // 8 waves * 1024 f32 (x tiles)
  const int tid  = threadIdx.x;
  const int wave = __builtin_amdgcn_readfirstlane(tid >> 5);  // uniform for TDM SGPRs
  const int lane = tid & 31;
  const int lr = lane & 15, hi = lane >> 4;

  const int n0 = blockIdx.x * 16 + wave * 2;
  float* xsf = sX + wave * 1024;                     // [t 0..7][node 0..1][c 0..63] f32

  // TDM: weight blob (1 x 24576 dwords) once per block; x tile (128 x 8, stride N*CI) per wave
  if (wave == 0)
    tdm_load_2d((u32)(uintptr_t)sB, blob, 24576u, 1u, 24576u, 1u, 24576u);
  tdm_load_2d((u32)(uintptr_t)xsf, x + (size_t)n0 * CI,
              (u32)(Nn * CI), (u32)Tt, 128u, (u32)Tt, (u32)(Nn * CI));
  if (tid < 128) zrow[tid] = 0.f;
  __builtin_amdgcn_s_wait_tensorcnt(0);
  __syncthreads();

  // A fragments: 3 temporal shifts x 2 K-tiles, f32 LDS -> f16 (ISA 16-bit A layout)
  v16h afr[6];
  {
    const int node = lr >> 3, t = lr & 7;
    #pragma unroll
    for (int s = 0; s < 3; ++s) {
      const int tt = t + s - 1;
      const float* base = (tt >= 0 && tt < Tt) ? &xsf[tt * 128 + node * 64]
                                               : (zrow + node * 64);
      #pragma unroll
      for (int kt = 0; kt < 2; ++kt) {
        const float* p = base + kt * 32 + hi * 8;
        v8f lo = *(const v8f*)p;
        v8f hv = *(const v8f*)(p + 16);
        v8h l16 = __builtin_convertvector(lo, v8h);
        v8h h16 = __builtin_convertvector(hv, v8h);
        afr[s * 2 + kt] = __builtin_shufflevector(l16, h16,
            0,1,2,3,4,5,6,7,8,9,10,11,12,13,14,15);
      }
    }
  }
  // accumulators seeded with conv bias
  v8f acc[8];
  #pragma unroll
  for (int ot = 0; ot < 8; ++ot) {
    float bv = tconv_b[ot * 16 + lr];
    #pragma unroll
    for (int f = 0; f < 8; ++f) acc[ot][f] = bv;
  }
  // temporal conv: 48 WMMAs
  #pragma unroll
  for (int ot = 0; ot < 8; ++ot)
    #pragma unroll
    for (int sk = 0; sk < 6; ++sk) {
      v16h b = *(const v16h*)&sB[(size_t)(sk * 8 + ot) * 512 + lane * 16];
      acc[ot] = WMMA16(afr[sk], b, acc[ot]);
    }
  // LayerNorm over 128 channels per row + ReLU
  float gg[8], bb[8];
  #pragma unroll
  for (int ot = 0; ot < 8; ++ot) { gg[ot] = ln_g[ot * 16 + lr]; bb[ot] = ln_b[ot * 16 + lr]; }
  #pragma unroll
  for (int f = 0; f < 8; ++f) {
    float s1 = 0.f, s2 = 0.f;
    #pragma unroll
    for (int ot = 0; ot < 8; ++ot) { float v = acc[ot][f]; s1 += v; s2 += v * v; }
    #pragma unroll
    for (int m = 1; m < 16; m <<= 1) {               // reduce within 16-lane half
      s1 += __shfl_xor(s1, m, 32);
      s2 += __shfl_xor(s2, m, 32);
    }
    float mu  = s1 * (1.f / 128.f);
    float var = s2 * (1.f / 128.f) - mu * mu;
    float rs  = rsqrtf(var + 1e-5f);
    #pragma unroll
    for (int ot = 0; ot < 8; ++ot) {
      float v = (acc[ot][f] - mu) * rs * gg[ot] + bb[ot];
      acc[ot][f] = fmaxf(v, 0.f);
    }
  }
  // residual: bias then 1x1 conv accumulated in-place (A = center shift frags)
  #pragma unroll
  for (int ot = 0; ot < 8; ++ot) {
    float rb = res_b[ot * 16 + lr];
    #pragma unroll
    for (int f = 0; f < 8; ++f) acc[ot][f] += rb;
    #pragma unroll
    for (int kt = 0; kt < 2; ++kt) {
      v16h b = *(const v16h*)&sB[(size_t)(48 + kt * 8 + ot) * 512 + lane * 16];
      acc[ot] = WMMA16(afr[2 + kt], b, acc[ot]);
    }
  }
  // stage xcmb (f16, 16x128 row-major) overlaying the dead x-tile region
  _Float16* st = (_Float16*)xsf;
  #pragma unroll
  for (int ot = 0; ot < 8; ++ot)
    #pragma unroll
    for (int f = 0; f < 8; ++f)
      st[(f + hi * 8) * 128 + ot * 16 + lr] = (_Float16)acc[ot][f];
  __syncthreads();
  v16h ag[4];
  #pragma unroll
  for (int kt = 0; kt < 4; ++kt) {
    const _Float16* p = &st[lr * 128 + kt * 32 + hi * 8];
    v8h lo = *(const v8h*)p;
    v8h hv = *(const v8h*)(p + 16);
    ag[kt] = __builtin_shufflevector(lo, hv, 0,1,2,3,4,5,6,7,8,9,10,11,12,13,14,15);
  }
  #pragma unroll
  for (int ot = 0; ot < 8; ++ot)
    #pragma unroll
    for (int f = 0; f < 8; ++f) acc[ot][f] = 0.f;
  #pragma unroll
  for (int ot = 0; ot < 8; ++ot)
    #pragma unroll
    for (int kt = 0; kt < 4; ++kt) {
      v16h b = *(const v16h*)&sB[(size_t)(64 + kt * 8 + ot) * 512 + lane * 16];
      acc[ot] = WMMA16(ag[kt], b, acc[ot]);
    }
  // store xt [n*8+t][128] f32 (L2-resident)
  const int rowbase = blockIdx.x * 128 + wave * 16;  // = n0*8
  #pragma unroll
  for (int ot = 0; ot < 8; ++ot)
    #pragma unroll
    for (int f = 0; f < 8; ++f)
      xt[(size_t)(rowbase + f + hi * 8) * CO + ot * 16 + lr] = acc[ot][f];
}

// ------------------------------------------------- edge message scatter
__global__ void k_scatter(const int* __restrict__ ei, const float* __restrict__ ew,
                          const float* __restrict__ dinv, const float* __restrict__ xt,
                          float* __restrict__ out) {
  const int e   = blockIdx.x;
  const int row = ei[e], col = ei[Ee + e];
  const float norm = dinv[row] * ew[e] * dinv[col];
  const int i = threadIdx.x;                         // 0..255, 4 floats each
  const float4 v = ((const float4*)xt)[row * 256 + i];
  const int j = i * 4, t = j >> 7, o = j & 127;
  float* op = out + ((size_t)t * Nn + col) * CO + o;
  atomicAdd(op + 0, norm * v.x);
  atomicAdd(op + 1, norm * v.y);
  atomicAdd(op + 2, norm * v.z);
  atomicAdd(op + 3, norm * v.w);
}

// ------------------------------------- self-loop + bias + ReLU (in place)
__global__ void k_final(float* __restrict__ out, const float* __restrict__ xt,
                        const float* __restrict__ dinv, const float* __restrict__ gcn_b) {
  long idx = (long)blockIdx.x * 256 + threadIdx.x;
  long j = idx * 4;
  if (j >= (long)Nn * Tt * CO) return;
  int o = (int)(j & 127);
  long r = j >> 7;                                   // t*Nn + n
  int n = (int)(r % Nn), t = (int)(r / Nn);
  float di = dinv[n]; float sn = di * di;            // self-loop norm
  float4 a = ((float4*)out)[idx];
  float4 xv = ((const float4*)xt)[((long)n * Tt + t) * 32 + (o >> 2)];
  float4 bv = *(const float4*)(gcn_b + o);
  a.x = fmaxf(a.x + sn * xv.x + bv.x, 0.f);
  a.y = fmaxf(a.y + sn * xv.y + bv.y, 0.f);
  a.z = fmaxf(a.z + sn * xv.z + bv.z, 0.f);
  a.w = fmaxf(a.w + sn * xv.w + bv.w, 0.f);
  ((float4*)out)[idx] = a;
}

extern "C" void kernel_launch(void* const* d_in, const int* in_sizes, int n_in,
                              void* d_out, int out_size, void* d_ws, size_t ws_size,
                              hipStream_t stream) {
  (void)in_sizes; (void)n_in; (void)ws_size;
  const float* x       = (const float*)d_in[0];
  const int*   ei      = (const int*)  d_in[1];
  const float* ew_in   = (const float*)d_in[2];
  const float* tconv_w = (const float*)d_in[3];
  const float* tconv_b = (const float*)d_in[4];
  const float* ln_g    = (const float*)d_in[5];
  const float* ln_b    = (const float*)d_in[6];
  const float* res_w   = (const float*)d_in[7];
  const float* res_b   = (const float*)d_in[8];
  const float* gcn_w   = (const float*)d_in[9];
  const float* gcn_b   = (const float*)d_in[10];

  float* ws   = (float*)d_ws;
  float* ew   = ws + OFF_EW;
  float* deg  = ws + OFF_DEG;
  float* sc   = ws + OFF_SC;
  _Float16* blob = (_Float16*)(ws + OFF_BLOB);
  float* xt   = ws + OFF_XT;
  float* out  = (float*)d_out;

  (void)hipMemsetAsync(d_out, 0, (size_t)out_size * sizeof(float), stream);
  k_init <<<(Nn + 255) / 256, 256, 0, stream>>>(deg, sc);
  k_wprep<<<192, 256, 0, stream>>>(tconv_w, res_w, gcn_w, blob);
  k_max  <<<256, 256, 0, stream>>>(ew_in, sc);
  k_sum  <<<256, 256, 0, stream>>>(ew_in, sc);
  k_edge <<<(Ee + 255) / 256, 256, 0, stream>>>(ew_in, ei, sc, ew, deg);
  k_dinv <<<(Nn + 255) / 256, 256, 0, stream>>>(deg);

  const int SMEM = 96 * 512 * 2 + 128 * 4 + 8 * 1024 * 4;   // 131584 B
  (void)hipFuncSetAttribute(reinterpret_cast<const void*>(kb_fused),
                            hipFuncAttributeMaxDynamicSharedMemorySize, SMEM);
  kb_fused<<<Nn / 16, 256, SMEM, stream>>>(x, blob, tconv_b, ln_g, ln_b, res_b, xt);

  k_scatter<<<Ee, 256, 0, stream>>>(ei, ew, deg, xt, out);
  k_final  <<<(Nn * Tt * CO / 4 + 255) / 256, 256, 0, stream>>>(out, xt, deg, gcn_b);
}
